// GRUGen_37108517438019
// MI455X (gfx1250) — matrix operand order repaired and checked by reading, not compiled, not run on hardware
//
#include <hip/hip_runtime.h>
#include <hip/hip_bf16.h>

// ---------------------------------------------------------------------------
// Types for CDNA5 WMMA (wave32, V_WMMA_F32_16X16X32_BF16)
// ---------------------------------------------------------------------------
typedef unsigned short u16;
typedef unsigned int   u32;
typedef __attribute__((ext_vector_type(16))) __bf16 v16bf;
typedef __attribute__((ext_vector_type(8)))  float  v8f;
typedef __attribute__((ext_vector_type(4)))  u32    v4u;
typedef __attribute__((ext_vector_type(8)))  u32    v8u;

#define T_STEPS 512
#define B_SZ    64
#define HE_SZ   512
#define HD_SZ   512
#define D_SZ    528
#define D_PAD   544
#define G3      1536

__device__ __forceinline__ u16 f2bf(float f) {
    u32 u = __builtin_bit_cast(u32, f);
    u += 0x7fffu + ((u >> 16) & 1u);   // round-to-nearest-even
    return (u16)(u >> 16);
}
__device__ __forceinline__ float bf2f(u16 h) {
    return __builtin_bit_cast(float, (u32)h << 16);
}
__device__ __forceinline__ float sigm(float x) { return 1.0f / (1.0f + __expf(-x)); }

// A fragment: 16x32 bf16 tile; lanes 0-15 row M=lane, K {0..15};
// lanes 16-31 same rows, K {16..31}. p already points at (row, lane K base).
__device__ __forceinline__ v16bf ld_a(const u16* __restrict__ p) {
    v4u lo = *(const v4u*)p;
    v4u hi = *(const v4u*)(p + 8);
    v8u raw;
    raw[0] = lo[0]; raw[1] = lo[1]; raw[2] = lo[2]; raw[3] = lo[3];
    raw[4] = hi[0]; raw[5] = hi[1]; raw[6] = hi[2]; raw[7] = hi[3];
    return __builtin_bit_cast(v16bf, raw);
}
// B fragment: 32x16 tile of W^T; lane holds W row n = n0+(lane&15),
// 16 contiguous K values starting at k0 + 16*(lane>>4). One 32B load.
__device__ __forceinline__ v16bf ld_b(const u16* __restrict__ p) {
    return __builtin_bit_cast(v16bf, *(const v8u*)p);
}

__device__ __forceinline__ void epilogue_tile(const v8f& acc, int rbase, int col, int N,
                                              const float* __restrict__ bias,
                                              const float* __restrict__ addend, int ladd,
                                              float* __restrict__ C, int ldc,
                                              u16* __restrict__ Cb, int ldcb, int row_mode) {
    bool colok = col < N;
    float bcol = (bias && colok) ? bias[col] : 0.0f;
    #pragma unroll
    for (int r = 0; r < 8; ++r) {
        int row = rbase + r;
        float v = acc[r] + bcol;
        if (addend && colok) v += addend[(size_t)row * ladd + col];
        int orow = row;
        if (row_mode == 1) {                  // rows are b*512+t; drop t==511
            int b_ = row >> 9, t_ = row & 511;
            if (t_ == T_STEPS - 1) continue;
            orow = b_ * (T_STEPS - 1) + t_;
        }
        if (colok) {
            if (C)  C[(size_t)orow * ldc + col] = v;
            if (Cb) Cb[(size_t)orow * ldcb + col] = f2bf(v);
        }
    }
}

// ---------------------------------------------------------------------------
// WMMA GEMM:  C[M,N] = A[M,K] @ W[Npad,K]^T  (+ bias[n]) (+ addend[m,n])
// Npad = 64*ngroups rows are physically allocated in W (zero padded), so every
// wave does exactly 4 unconditional 16x16 tiles -> branch-free K loop:
// 2x global_load_b128 (A) + 4x 32B loads (B) + 4x v_wmma.
// All tile indexing is SGPR-uniform (readfirstlane): EXEC all-ones at WMMAs.
// ---------------------------------------------------------------------------
__global__ void wmma_gemm_kernel(const u16* __restrict__ A, int lda,
                                 const u16* __restrict__ W, int ldw,
                                 const float* __restrict__ bias,
                                 const float* __restrict__ addend, int ladd,
                                 float* __restrict__ C, int ldc,
                                 u16* __restrict__ Cb, int ldcb,
                                 int M, int N, int ngroups, int K, int row_mode) {
    int wave = __builtin_amdgcn_readfirstlane((int)(threadIdx.x >> 5));
    int lane = threadIdx.x & 31;
    int mtiles = M >> 4;
    int unit = blockIdx.x * 8 + wave;                 // SGPR-uniform
    if (unit >= ngroups * mtiles) return;             // scalar branch
    int mt = unit / ngroups;
    int ng = unit - mt * ngroups;
    int row0 = mt << 4;
    int n0 = ng << 6;

    const u16* Ap = A + (size_t)(row0 + (lane & 15)) * lda + ((lane >> 4) << 4);
    const u16* Bp = W + (size_t)(n0 + (lane & 15)) * ldw + ((lane >> 4) << 4);
    const size_t bstep = (size_t)16 * ldw;

    v8f acc0 = {}, acc1 = {}, acc2 = {}, acc3 = {};
    #pragma clang loop unroll(disable)
    for (int k0 = 0; k0 < K; k0 += 32) {
        v16bf a  = ld_a(Ap + k0);
        v16bf b0 = ld_b(Bp + k0);
        v16bf b1 = ld_b(Bp + bstep + k0);
        v16bf b2 = ld_b(Bp + 2 * bstep + k0);
        v16bf b3 = ld_b(Bp + 3 * bstep + k0);
        acc0 = __builtin_amdgcn_wmma_f32_16x16x32_bf16(false, a, false, b0, (short)0, acc0, false, false);
        acc1 = __builtin_amdgcn_wmma_f32_16x16x32_bf16(false, a, false, b1, (short)0, acc1, false, false);
        acc2 = __builtin_amdgcn_wmma_f32_16x16x32_bf16(false, a, false, b2, (short)0, acc2, false, false);
        acc3 = __builtin_amdgcn_wmma_f32_16x16x32_bf16(false, a, false, b3, (short)0, acc3, false, false);
    }

    int rbase = row0 + ((lane >> 4) << 3);
    int colb  = n0 + (lane & 15);
    epilogue_tile(acc0, rbase, colb +  0, N, bias, addend, ladd, C, ldc, Cb, ldcb, row_mode);
    epilogue_tile(acc1, rbase, colb + 16, N, bias, addend, ladd, C, ldc, Cb, ldcb, row_mode);
    epilogue_tile(acc2, rbase, colb + 32, N, bias, addend, ladd, C, ldc, Cb, ldcb, row_mode);
    epilogue_tile(acc3, rbase, colb + 48, N, bias, addend, ladd, C, ldc, Cb, ldcb, row_mode);
}

// f32 -> bf16 with zero padding in rows and columns
__global__ void cvt_pad_kernel(const float* __restrict__ src, u16* __restrict__ dst,
                               int srows, int scols, int drows, int dcols) {
    size_t i = (size_t)blockIdx.x * 256 + threadIdx.x;
    if (i >= (size_t)drows * dcols) return;
    int r = (int)(i / dcols), c = (int)(i % dcols);
    dst[i] = (r < srows && c < scols) ? f2bf(src[(size_t)r * scols + c]) : (u16)0;
}

// Wemb (8,500,64) f32 -> (8,512,64) bf16, vocab rows 500..511 zero-padded
__global__ void cvt_emb_kernel(const float* __restrict__ src, u16* __restrict__ dst) {
    int i = blockIdx.x * 256 + threadIdx.x;           // 8*512*64 = 262144
    if (i >= 8 * 512 * 64) return;
    int f = i & 63;
    int v = (i >> 6) & 511;
    int e = i >> 15;
    dst[i] = (v < 500) ? f2bf(src[((e * 500 + v) << 6) + f]) : (u16)0;
}

// Encoder GRU gate update + last_hidden capture
__global__ void enc_step_pw(const float* __restrict__ xg, int t,
                            const float* __restrict__ hg,
                            float* __restrict__ h, u16* __restrict__ hb,
                            const int* __restrict__ seq_lens,
                            float* __restrict__ last_h) {
    int i = blockIdx.x * 256 + threadIdx.x;
    if (i >= B_SZ * HE_SZ) return;
    int b = i >> 9, j = i & 511;
    const float* xr = xg + ((size_t)b * T_STEPS + t) * G3;
    const float* hr = hg + (size_t)b * G3;
    float r = sigm(xr[j] + hr[j]);
    float z = sigm(xr[HE_SZ + j] + hr[HE_SZ + j]);
    float n = tanhf(xr[2 * HE_SZ + j] + r * hr[2 * HE_SZ + j]);
    float hn = z * h[i] + (1.0f - z) * n;
    h[i] = hn;
    hb[i] = f2bf(hn);
    int lt = seq_lens[b] - 1;
    lt = lt < 0 ? 0 : (lt > T_STEPS - 1 ? T_STEPS - 1 : lt);
    if (t == lt) last_h[i] = hn;
}

// Decoder GRU gate update: hnew = z*hx + (1-z)*tanh(xn + r*hn)
__global__ void dec_step_pw(const float* __restrict__ xg, size_t xg_stride,
                            const float* __restrict__ hg,
                            const float* __restrict__ hx,
                            float* __restrict__ h, u16* __restrict__ hb,
                            u16* __restrict__ dec_slot, size_t dec_stride) {
    int i = blockIdx.x * 256 + threadIdx.x;
    if (i >= B_SZ * HD_SZ) return;
    int b = i >> 9, j = i & 511;
    const float* xr = xg + (size_t)b * xg_stride;
    const float* hr = hg + (size_t)b * G3;
    float hxv = hx[i];
    float r = sigm(xr[j] + hr[j]);
    float z = sigm(xr[HD_SZ + j] + hr[HD_SZ + j]);
    float n = tanhf(xr[2 * HD_SZ + j] + r * hr[2 * HD_SZ + j]);
    float hn = z * hxv + (1.0f - z) * n;
    h[i] = hn;
    u16 hv = f2bf(hn);
    hb[i] = hv;
    if (dec_slot) dec_slot[(size_t)b * dec_stride + j] = hv;
}

// pred_delta = dec[:, :-1] @ Wdel^T + bdel  (N=1 GEMV, tiny)
__global__ void delta_kernel(const u16* __restrict__ decb,
                             const float* __restrict__ Wdel,
                             const float* __restrict__ bdel,
                             float* __restrict__ outd) {
    int i = blockIdx.x * 256 + threadIdx.x;
    if (i >= B_SZ * T_STEPS) return;
    int b = i >> 9, t = i & 511;
    if (t == T_STEPS - 1) return;
    const u16* row = decb + (size_t)i * HD_SZ;
    float s = bdel[0];
    for (int k = 0; k < HD_SZ; ++k) s += bf2f(row[k]) * Wdel[k];
    outd[b * (T_STEPS - 1) + t] = s;
}

__global__ void copy_f32(const float* __restrict__ s, float* __restrict__ d, int n) {
    int i = blockIdx.x * 256 + threadIdx.x;
    if (i < n) d[i] = s[i];
}

// ---------------------------------------------------------------------------
extern "C" void kernel_launch(void* const* d_in, const int* in_sizes, int n_in,
                              void* d_out, int out_size, void* d_ws, size_t ws_size,
                              hipStream_t stream) {
    (void)in_sizes; (void)n_in; (void)out_size; (void)ws_size;

    const float* x    = (const float*)d_in[0];
    const float* Wih  = (const float*)d_in[1];
    const float* Whh  = (const float*)d_in[2];
    const float* bih  = (const float*)d_in[3];
    const float* bhh  = (const float*)d_in[4];
    const float* W0x  = (const float*)d_in[5];
    const float* b0x  = (const float*)d_in[6];
    const float* W0h  = (const float*)d_in[7];
    const float* b0h  = (const float*)d_in[8];
    const float* W0m  = (const float*)d_in[9];
    const float* b0m  = (const float*)d_in[10];
    const float* W1x  = (const float*)d_in[11];
    const float* b1x  = (const float*)d_in[12];
    const float* W1h  = (const float*)d_in[13];
    const float* b1h  = (const float*)d_in[14];
    const float* W1m  = (const float*)d_in[15];
    const float* b1m  = (const float*)d_in[16];
    const float* Wout = (const float*)d_in[17];
    const float* bout = (const float*)d_in[18];
    const float* Wdel = (const float*)d_in[19];
    const float* bdel = (const float*)d_in[20];
    const float* Wemb = (const float*)d_in[21];
    const float* bemb = (const float*)d_in[22];
    const int* seq_lens = (const int*)d_in[23];

    const size_t BT = (size_t)B_SZ * T_STEPS;          // 32768
    const size_t M_OUT = (size_t)B_SZ * (T_STEPS - 1); // 32704

    float* out_out = (float*)d_out;                    // (B, 511, 528)
    float* out_del = out_out + M_OUT * D_SZ;           // (B, 511)
    float* out_emb = out_del + M_OUT;                  // (8, B, 511, 500)
    float* out_lh  = out_emb + (size_t)8 * M_OUT * 500;// (B, 512)

    char* w = (char*)d_ws;
    size_t off = 0;
    auto alloc = [&](size_t bytes) -> void* {
        void* p = w + off;
        off = (off + bytes + 255) & ~(size_t)255;
        return p;
    };

    u16* xb    = (u16*)alloc(BT * D_PAD * 2);
    u16* Wihb  = (u16*)alloc((size_t)G3 * D_PAD * 2);
    u16* W0xb  = (u16*)alloc((size_t)G3 * D_PAD * 2);
    u16* Whhb  = (u16*)alloc((size_t)G3 * HE_SZ * 2);
    u16* W0hb  = (u16*)alloc((size_t)G3 * HD_SZ * 2);
    u16* W1xb  = (u16*)alloc((size_t)G3 * HD_SZ * 2);
    u16* W1hb  = (u16*)alloc((size_t)G3 * HD_SZ * 2);
    u16* W0mb  = (u16*)alloc((size_t)HD_SZ * 1024 * 2);
    u16* W1mb  = (u16*)alloc((size_t)HD_SZ * 1024 * 2);
    u16* Woutb = (u16*)alloc((size_t)576 * HD_SZ * 2);  // rows padded 528->576
    u16* Wembb = (u16*)alloc((size_t)8 * 512 * 64 * 2); // vocab padded 500->512

    float* xgE  = (float*)alloc(BT * G3 * 4);
    float* x2h  = (float*)alloc(BT * G3 * 4);
    float* hgE  = (float*)alloc((size_t)B_SZ * G3 * 4);
    float* hg0  = (float*)alloc((size_t)B_SZ * G3 * 4);
    float* xg1  = (float*)alloc((size_t)B_SZ * G3 * 4);
    float* hg1  = (float*)alloc((size_t)B_SZ * G3 * 4);
    float* hx0  = (float*)alloc((size_t)B_SZ * HD_SZ * 4);
    float* hx1  = (float*)alloc((size_t)B_SZ * HD_SZ * 4);
    float* ghm0 = (float*)alloc((size_t)B_SZ * HD_SZ * 4);
    float* ghm1 = (float*)alloc((size_t)B_SZ * HD_SZ * 4);
    float* hE   = (float*)alloc((size_t)B_SZ * HE_SZ * 4);
    float* h0   = (float*)alloc((size_t)B_SZ * HD_SZ * 4);
    float* h1   = (float*)alloc((size_t)B_SZ * HD_SZ * 4);
    float* lhf  = (float*)alloc((size_t)B_SZ * HE_SZ * 4);

    u16* hEb  = (u16*)alloc((size_t)B_SZ * HE_SZ * 2);
    u16* h0b  = (u16*)alloc((size_t)B_SZ * HD_SZ * 2);
    u16* h1b  = (u16*)alloc((size_t)B_SZ * HD_SZ * 2);
    u16* lhb  = (u16*)alloc((size_t)B_SZ * HE_SZ * 2);
    u16* hx0b = (u16*)alloc((size_t)B_SZ * HD_SZ * 2);
    u16* hx1b = (u16*)alloc((size_t)B_SZ * HD_SZ * 2);
    u16* decb = (u16*)alloc(BT * HD_SZ * 2);
    u16* outb = (u16*)alloc(M_OUT * D_SZ * 2);

    auto gemm = [&](const u16* A, int lda, const u16* W_, int ldw,
                    const float* bias, const float* add, int ladd,
                    float* C, int ldc, u16* Cb, int ldcb,
                    int M, int N, int Npad, int K, int mode) {
        int ngroups = Npad >> 6;                       // Npad always multiple of 64
        int units = (M >> 4) * ngroups;
        int blocks = (units + 7) / 8;
        wmma_gemm_kernel<<<blocks, 256, 0, stream>>>(A, lda, W_, ldw, bias, add, ladd,
                                                     C, ldc, Cb, ldcb, M, N, ngroups, K, mode);
    };
    auto cvt = [&](const float* s, u16* d, int srows, int scols, int drows, int dcols) {
        size_t tot = (size_t)drows * dcols;
        cvt_pad_kernel<<<(int)((tot + 255) / 256), 256, 0, stream>>>(s, d, srows, scols, drows, dcols);
    };

    // ---- weight / input quantization (padded where needed) ----
    cvt(x,    xb,    (int)BT, D_SZ, (int)BT, D_PAD);
    cvt(Wih,  Wihb,  G3, D_SZ, G3, D_PAD);
    cvt(W0x,  W0xb,  G3, D_SZ, G3, D_PAD);
    cvt(Whh,  Whhb,  G3, HE_SZ, G3, HE_SZ);
    cvt(W0h,  W0hb,  G3, HD_SZ, G3, HD_SZ);
    cvt(W1x,  W1xb,  G3, HD_SZ, G3, HD_SZ);
    cvt(W1h,  W1hb,  G3, HD_SZ, G3, HD_SZ);
    cvt(W0m,  W0mb,  HD_SZ, 1024, HD_SZ, 1024);
    cvt(W1m,  W1mb,  HD_SZ, 1024, HD_SZ, 1024);
    cvt(Wout, Woutb, D_SZ, HD_SZ, 576, HD_SZ);
    cvt_emb_kernel<<<(8 * 512 * 64) / 256, 256, 0, stream>>>(Wemb, Wembb);

    hipMemsetAsync(hE,  0, (size_t)B_SZ * HE_SZ * 4, stream);
    hipMemsetAsync(h0,  0, (size_t)B_SZ * HD_SZ * 4, stream);
    hipMemsetAsync(h1,  0, (size_t)B_SZ * HD_SZ * 4, stream);
    hipMemsetAsync(hEb, 0, (size_t)B_SZ * HE_SZ * 2, stream);
    hipMemsetAsync(h0b, 0, (size_t)B_SZ * HD_SZ * 2, stream);
    hipMemsetAsync(h1b, 0, (size_t)B_SZ * HD_SZ * 2, stream);

    // ---- big input projections ----
    gemm(xb, D_PAD, Wihb, D_PAD, bih, nullptr, 0, xgE, G3, nullptr, 0, (int)BT, G3, G3, D_PAD, 0);
    gemm(xb, D_PAD, W0xb, D_PAD, b0x, nullptr, 0, x2h, G3, nullptr, 0, (int)BT, G3, G3, D_PAD, 0);

    // ---- encoder scan ----
    for (int t = 0; t < T_STEPS; ++t) {
        gemm(hEb, HE_SZ, Whhb, HE_SZ, bhh, nullptr, 0, hgE, G3, nullptr, 0,
             B_SZ, G3, G3, HE_SZ, 0);
        enc_step_pw<<<(B_SZ * HE_SZ) / 256, 256, 0, stream>>>(xgE, t, hgE, hE, hEb, seq_lens, lhf);
    }
    cvt(lhf, lhb, B_SZ, HE_SZ, B_SZ, HE_SZ);
    gemm(lhb, HE_SZ, W0mb, 1024, b0m, nullptr, 0, ghm0, HD_SZ, nullptr, 0,
         B_SZ, HD_SZ, HD_SZ, HE_SZ, 0);
    gemm(lhb, HE_SZ, W1mb, 1024, b1m, nullptr, 0, ghm1, HD_SZ, nullptr, 0,
         B_SZ, HD_SZ, HD_SZ, HE_SZ, 0);

    // ---- decoder scan (2-layer GRU) ----
    for (int t = 0; t < T_STEPS; ++t) {
        // hx0 = ghm0 + h0 @ W0m[:,HE:]^T   (bf16 mirror for next GEMM)
        gemm(h0b, HD_SZ, W0mb + HE_SZ, 1024, nullptr, ghm0, HD_SZ, hx0, HD_SZ, hx0b, HD_SZ,
             B_SZ, HD_SZ, HD_SZ, HD_SZ, 0);
        // hg0 = hx0 @ W0h^T + b0h
        gemm(hx0b, HD_SZ, W0hb, HD_SZ, b0h, nullptr, 0, hg0, G3, nullptr, 0,
             B_SZ, G3, G3, HD_SZ, 0);
        // h0 = gru(x2h0[t], hg0, hx0)
        dec_step_pw<<<(B_SZ * HD_SZ) / 256, 256, 0, stream>>>(
            x2h + (size_t)t * G3, (size_t)T_STEPS * G3, hg0, hx0, h0, h0b, (u16*)nullptr, 0);
        // xg1 = h0 @ W1x^T + b1x
        gemm(h0b, HD_SZ, W1xb, HD_SZ, b1x, nullptr, 0, xg1, G3, nullptr, 0,
             B_SZ, G3, G3, HD_SZ, 0);
        // hx1 = ghm1 + h1 @ W1m[:,HE:]^T
        gemm(h1b, HD_SZ, W1mb + HE_SZ, 1024, nullptr, ghm1, HD_SZ, hx1, HD_SZ, hx1b, HD_SZ,
             B_SZ, HD_SZ, HD_SZ, HD_SZ, 0);
        // hg1 = hx1 @ W1h^T + b1h
        gemm(hx1b, HD_SZ, W1hb, HD_SZ, b1h, nullptr, 0, hg1, G3, nullptr, 0,
             B_SZ, G3, G3, HD_SZ, 0);
        // h1 = gru(xg1, hg1, hx1); also record dec[b, t, :]
        dec_step_pw<<<(B_SZ * HD_SZ) / 256, 256, 0, stream>>>(
            xg1, (size_t)G3, hg1, hx1, h1, h1b,
            decb + (size_t)t * HD_SZ, (size_t)T_STEPS * HD_SZ);
    }

    // ---- outputs ----
    // out = dec @ Wout^T + bout, dropping t == T-1 (row_mode 1), bf16 mirror for emb GEMM
    gemm(decb, HD_SZ, Woutb, HD_SZ, bout, nullptr, 0, out_out, D_SZ, outb, D_SZ,
         (int)BT, D_SZ, 576, HD_SZ, 1);
    // pred_delta
    delta_kernel<<<(int)((BT + 255) / 256), 256, 0, stream>>>(decb, Wdel, bdel, out_del);
    // emb_logits[e] = resized[:,:,e,:] @ Wemb[e]^T + bemb[e]  (vocab padded to 512)
    for (int e = 0; e < 8; ++e) {
        gemm(outb + e * 64, D_SZ, Wembb + (size_t)e * 512 * 64, 64, bemb + e * 500,
             nullptr, 0, out_emb + (size_t)e * M_OUT * 500, 500, nullptr, 0,
             (int)M_OUT, 500, 512, 64, 0);
    }
    // last_hidden
    copy_f32<<<(B_SZ * HE_SZ) / 256, 256, 0, stream>>>(lhf, out_lh, B_SZ * HE_SZ);
}
